// Transducer_19585050870112
// MI455X (gfx1250) — matrix operand order repaired and checked by reading, not compiled, not run on hardware
//
#include <hip/hip_runtime.h>
#include <hip/hip_bf16.h>
#include <math.h>

// ---------------------------------------------------------------------------
// Problem constants (from reference): B=4, T=256, U=64, D=512, K=512, V=1000
// ---------------------------------------------------------------------------
#define DIM_B 4
#define DIM_T 256
#define DIM_U 64
#define DIM_D 512
#define DIM_K 512
#define DIM_V 1000

#define KSTEPS (DIM_K / 32)        // 16 K-steps of 32 for wmma 16x16x32
#define NT_K   (DIM_K / 16)        // 32 column tiles for stage-1 GEMMs
#define NT_V   ((DIM_V + 15) / 16) // 63 column tiles for joint GEMM
#define RT_J   4                   // row-tiles per block in joint kernel (64 rows)

typedef __attribute__((ext_vector_type(16))) __bf16 v16bf;
typedef __attribute__((ext_vector_type(8)))  float  v8f;

// ---------------------------------------------------------------------------
// bf16 split helpers: x ~= hi + lo, each bf16 (round-to-nearest-even).
// hi+lo captures ~16 mantissa bits -> 3-product WMMA gives near-f32 GEMM.
// ---------------------------------------------------------------------------
__device__ __forceinline__ unsigned short f32_to_bf16_rne(float f) {
    union { float f; unsigned int u; } x; x.f = f;
    unsigned int u = x.u;
    u += 0x7FFFu + ((u >> 16) & 1u);
    return (unsigned short)(u >> 16);
}
__device__ __forceinline__ float bf16_bits_to_f32(unsigned short h) {
    union { unsigned int u; float f; } x; x.u = ((unsigned int)h) << 16;
    return x.f;
}
__device__ __forceinline__ void split_bf16(float f, unsigned short& hi, unsigned short& lo) {
    hi = f32_to_bf16_rne(f);
    lo = f32_to_bf16_rne(f - bf16_bits_to_f32(hi));
}

// ---------------------------------------------------------------------------
// Pack 8 f32 values (k0..k0+7, k0 % 8 == 0) for (rowTile rt, row-in-tile row16)
// into the ISA 16-bit A-fragment LDS layout with ONE 16B store per matrix.
// Layout per fragment lane (16 bf16): e = (K&7) + 8*((K>>4)&1),
// lane = row + 16*((K>>3)&1); an aligned 8-span of K is contiguous in e.
// ---------------------------------------------------------------------------
__device__ __forceinline__ void store_a_span8(v16bf* ldsHi, v16bf* ldsLo,
                                              int rt, int row16, int k0,
                                              const float v[8]) {
    unsigned int ph[4], pl[4];
#pragma unroll
    for (int i = 0; i < 4; ++i) {
        unsigned short h0, l0, h1, l1;
        split_bf16(v[2 * i], h0, l0);
        split_bf16(v[2 * i + 1], h1, l1);
        ph[i] = (unsigned int)h0 | ((unsigned int)h1 << 16);
        pl[i] = (unsigned int)l0 | ((unsigned int)l1 << 16);
    }
    int kstep = k0 >> 5;
    int q0 = k0 & 31;
    int lane = row16 + (((q0 >> 3) & 1) << 4);
    int u4idx = ((rt * KSTEPS + kstep) * 32 + lane) * 2 + ((q0 >> 4) & 1);
    ((uint4*)ldsHi)[u4idx] = make_uint4(ph[0], ph[1], ph[2], ph[3]);
    ((uint4*)ldsLo)[u4idx] = make_uint4(pl[0], pl[1], pl[2], pl[3]);
}

// ---------------------------------------------------------------------------
// Pre-swizzle a weight matrix into WMMA B-fragment layout (hi/lo bf16).
// Logical GEMM operand: Bmat[k][n] = W[n][colOff + k]   (n < nValid, else 0)
// Fragment layout (per ISA 16-bit B, 32x16 per kstep):
//   frag[(ntile*KSTEPS + kstep)*32 + lane][i] : K = kstep*32 + (lane>>4)*8 + i + 8*(i>>3)
//                                               N = ntile*16 + (lane&15)
// ---------------------------------------------------------------------------
__global__ void prep_b_frags(const float* __restrict__ W, int rowStride, int colOff,
                             int nValid, int nTiles,
                             unsigned short* __restrict__ hi,
                             unsigned short* __restrict__ lo) {
    int gid = blockIdx.x * blockDim.x + threadIdx.x;
    int total = nTiles * KSTEPS * 32;
    if (gid >= total) return;
    int lane  = gid & 31;
    int kstep = (gid >> 5) & (KSTEPS - 1);
    int nt    = gid >> 9;               // gid / (KSTEPS*32)
    int n     = nt * 16 + (lane & 15);
    int kbase = kstep * 32 + ((lane >> 4) * 8);
    unsigned short* dhi = hi + (size_t)gid * 16;
    unsigned short* dlo = lo + (size_t)gid * 16;
#pragma unroll
    for (int i = 0; i < 16; ++i) {
        float w = (n < nValid) ? W[(size_t)n * rowStride + colOff + kbase + i + ((i >> 3) << 3)] : 0.0f;
        unsigned short h, l; split_bf16(w, h, l);
        dhi[i] = h; dlo[i] = l;
    }
}

// ---------------------------------------------------------------------------
// Stage-1 GEMM: out[M x 512] = A[M x 512] * Bfrags (+ bias).  16 rows/block,
// 8 waves; wave w handles column tiles w, w+8, ... (32 tiles total).
// 3 WMMAs per K-step: hi*hi + hi*lo + lo*hi.
// ---------------------------------------------------------------------------
__global__ __launch_bounds__(256) void gemm_h_stage(
    const float* __restrict__ A,
    const v16bf* __restrict__ bhi, const v16bf* __restrict__ blo,
    const float* __restrict__ bias,      // nullable (length 512)
    float* __restrict__ out) {
    __shared__ v16bf ldsA_hi[KSTEPS * 32];
    __shared__ v16bf ldsA_lo[KSTEPS * 32];
    const int tid = threadIdx.x;
    const int r0  = blockIdx.x * 16;

    for (int idx = tid; idx < (16 * DIM_K) / 8; idx += 256) {
        int row = idx >> 6;                 // 64 chunks of 8 per row
        int k0  = (idx & 63) << 3;
        const float* src = A + (size_t)(r0 + row) * DIM_K + k0;
        float4 a0 = *(const float4*)(src);
        float4 a1 = *(const float4*)(src + 4);
        float v[8] = {a0.x, a0.y, a0.z, a0.w, a1.x, a1.y, a1.z, a1.w};
        store_a_span8(ldsA_hi, ldsA_lo, 0, row, k0, v);
    }
    __syncthreads();

    const int lane = tid & 31;
    const int wave = tid >> 5;
    for (int nt = wave; nt < NT_K; nt += 8) {
        v8f acc = {};
        for (int ks = 0; ks < KSTEPS; ++ks) {
            v16bf ah = ldsA_hi[ks * 32 + lane];
            v16bf al = ldsA_lo[ks * 32 + lane];
            v16bf bh = bhi[(size_t)(nt * KSTEPS + ks) * 32 + lane];
            v16bf bl = blo[(size_t)(nt * KSTEPS + ks) * 32 + lane];
            acc = __builtin_amdgcn_wmma_f32_16x16x32_bf16(false, ah, false, bh, (short)0, acc, false, false);
            acc = __builtin_amdgcn_wmma_f32_16x16x32_bf16(false, ah, false, bl, (short)0, acc, false, false);
            acc = __builtin_amdgcn_wmma_f32_16x16x32_bf16(false, al, false, bh, (short)0, acc, false, false);
        }
        int n    = lane & 15;
        int mhi  = (lane >> 4) * 8;     // D layout: VGPR j -> M = j + 8*(lane>>4)
        int ncol = nt * 16 + n;
        float badd = bias ? bias[ncol] : 0.0f;
#pragma unroll
        for (int j = 0; j < 8; ++j)
            out[(size_t)(r0 + mhi + j) * DIM_K + ncol] = acc[j] + badd;
    }
}

// ---------------------------------------------------------------------------
// Fused joint kernel: one block per (b, t) pair -> 64 rows (all of U),
// i.e. RT_J = 4 row-tiles, 8 waves, 256 threads, 1024 blocks.
// Phase 1: h = tanh(he[b,t,:] + hd'[b,u,:]) (hd' has b1 folded in), split to
//          hi/lo bf16 in swizzled LDS (128 KB) -- computed ONCE per block.
// Phase 2: each wave sweeps V-tiles (stride 8) with 4 accumulators, reusing
//          every B fragment 4x (cuts W2-fragment L2 traffic 4x vs 16-row
//          blocks); 16 K-steps x 4 row-tiles x 3 WMMAs; adds b2; stores f32.
// ---------------------------------------------------------------------------
__global__ __launch_bounds__(256) void joint_kernel(
    const float* __restrict__ he,       // (B*T) x 512
    const float* __restrict__ hd,       // (B*U) x 512 (includes b1)
    const v16bf* __restrict__ w2hi, const v16bf* __restrict__ w2lo,
    const float* __restrict__ b2,
    float* __restrict__ out) {
    __shared__ v16bf ldsA_hi[RT_J * KSTEPS * 32];   // 64 KB
    __shared__ v16bf ldsA_lo[RT_J * KSTEPS * 32];   // 64 KB
    const int tid = threadIdx.x;
    const int bt  = blockIdx.x;             // b*T + t
    const int b   = bt >> 8;
    const int t   = bt & (DIM_T - 1);
    const float* heRow  = he + (size_t)bt * DIM_K;
    const float* hdBase = hd + (size_t)(b * DIM_U) * DIM_K;

    for (int idx = tid; idx < (DIM_U * DIM_K) / 8; idx += 256) {
        int row = idx >> 6;                 // u in 0..63
        int k0  = (idx & 63) << 3;
        const float4 e0 = *(const float4*)(heRow + k0);
        const float4 e1 = *(const float4*)(heRow + k0 + 4);
        const float* hdRow = hdBase + (size_t)row * DIM_K + k0;
        const float4 d0 = *(const float4*)(hdRow);
        const float4 d1 = *(const float4*)(hdRow + 4);
        float v[8] = {tanhf(e0.x + d0.x), tanhf(e0.y + d0.y),
                      tanhf(e0.z + d0.z), tanhf(e0.w + d0.w),
                      tanhf(e1.x + d1.x), tanhf(e1.y + d1.y),
                      tanhf(e1.z + d1.z), tanhf(e1.w + d1.w)};
        store_a_span8(ldsA_hi, ldsA_lo, row >> 4, row & 15, k0, v);
    }
    __syncthreads();

    const int lane = tid & 31;
    const int wave = tid >> 5;
    const size_t rowBase = (size_t)bt * DIM_U;
    for (int nt = wave; nt < NT_V; nt += 8) {
        v8f acc0 = {}, acc1 = {}, acc2 = {}, acc3 = {};
        for (int ks = 0; ks < KSTEPS; ++ks) {
            v16bf bh = w2hi[(size_t)(nt * KSTEPS + ks) * 32 + lane];
            v16bf bl = w2lo[(size_t)(nt * KSTEPS + ks) * 32 + lane];
            v16bf ah, al;
            ah = ldsA_hi[(0 * KSTEPS + ks) * 32 + lane];
            al = ldsA_lo[(0 * KSTEPS + ks) * 32 + lane];
            acc0 = __builtin_amdgcn_wmma_f32_16x16x32_bf16(false, ah, false, bh, (short)0, acc0, false, false);
            acc0 = __builtin_amdgcn_wmma_f32_16x16x32_bf16(false, ah, false, bl, (short)0, acc0, false, false);
            acc0 = __builtin_amdgcn_wmma_f32_16x16x32_bf16(false, al, false, bh, (short)0, acc0, false, false);
            ah = ldsA_hi[(1 * KSTEPS + ks) * 32 + lane];
            al = ldsA_lo[(1 * KSTEPS + ks) * 32 + lane];
            acc1 = __builtin_amdgcn_wmma_f32_16x16x32_bf16(false, ah, false, bh, (short)0, acc1, false, false);
            acc1 = __builtin_amdgcn_wmma_f32_16x16x32_bf16(false, ah, false, bl, (short)0, acc1, false, false);
            acc1 = __builtin_amdgcn_wmma_f32_16x16x32_bf16(false, al, false, bh, (short)0, acc1, false, false);
            ah = ldsA_hi[(2 * KSTEPS + ks) * 32 + lane];
            al = ldsA_lo[(2 * KSTEPS + ks) * 32 + lane];
            acc2 = __builtin_amdgcn_wmma_f32_16x16x32_bf16(false, ah, false, bh, (short)0, acc2, false, false);
            acc2 = __builtin_amdgcn_wmma_f32_16x16x32_bf16(false, ah, false, bl, (short)0, acc2, false, false);
            acc2 = __builtin_amdgcn_wmma_f32_16x16x32_bf16(false, al, false, bh, (short)0, acc2, false, false);
            ah = ldsA_hi[(3 * KSTEPS + ks) * 32 + lane];
            al = ldsA_lo[(3 * KSTEPS + ks) * 32 + lane];
            acc3 = __builtin_amdgcn_wmma_f32_16x16x32_bf16(false, ah, false, bh, (short)0, acc3, false, false);
            acc3 = __builtin_amdgcn_wmma_f32_16x16x32_bf16(false, ah, false, bl, (short)0, acc3, false, false);
            acc3 = __builtin_amdgcn_wmma_f32_16x16x32_bf16(false, al, false, bh, (short)0, acc3, false, false);
        }
        int n    = lane & 15;
        int mhi  = (lane >> 4) * 8;         // D layout: VGPR j -> M = j + 8*(lane>>4)
        int ncol = nt * 16 + n;
        if (ncol < DIM_V) {
            float badd = b2[ncol];
            v8f accs[RT_J] = {acc0, acc1, acc2, acc3};
#pragma unroll
            for (int rt = 0; rt < RT_J; ++rt) {
#pragma unroll
                for (int j = 0; j < 8; ++j) {
                    int u = rt * 16 + mhi + j;
                    out[(rowBase + u) * DIM_V + ncol] = accs[rt][j] + badd;
                }
            }
        }
    }
}

// ---------------------------------------------------------------------------
// Workspace layout (bytes, all 32B-aligned)
// ---------------------------------------------------------------------------
static constexpr size_t HE_BYTES   = (size_t)DIM_B * DIM_T * DIM_K * 4;      // 2 MB
static constexpr size_t HD_BYTES   = (size_t)DIM_B * DIM_U * DIM_K * 4;      // 512 KB
static constexpr size_t W1F_BYTES  = (size_t)NT_K * KSTEPS * 32 * 16 * 2;    // 512 KB each
static constexpr size_t W2F_BYTES  = (size_t)NT_V * KSTEPS * 32 * 16 * 2;    // 1008 KB each

static constexpr size_t OFF_HE     = 0;
static constexpr size_t OFF_HD     = OFF_HE + HE_BYTES;
static constexpr size_t OFF_W1E_HI = OFF_HD + HD_BYTES;
static constexpr size_t OFF_W1E_LO = OFF_W1E_HI + W1F_BYTES;
static constexpr size_t OFF_W1D_HI = OFF_W1E_LO + W1F_BYTES;
static constexpr size_t OFF_W1D_LO = OFF_W1D_HI + W1F_BYTES;
static constexpr size_t OFF_W2_HI  = OFF_W1D_LO + W1F_BYTES;
static constexpr size_t OFF_W2_LO  = OFF_W2_HI + W2F_BYTES;

extern "C" void kernel_launch(void* const* d_in, const int* in_sizes, int n_in,
                              void* d_out, int out_size, void* d_ws, size_t ws_size,
                              hipStream_t stream) {
    const float* enc = (const float*)d_in[0];   // (4,256,512)
    const float* dec = (const float*)d_in[1];   // (4,64,512)
    const float* W1  = (const float*)d_in[2];   // (512,1024)
    const float* b1  = (const float*)d_in[3];   // (512,)
    const float* W2  = (const float*)d_in[4];   // (1000,512)
    const float* b2  = (const float*)d_in[5];   // (1000,)
    float* out = (float*)d_out;

    char* ws = (char*)d_ws;
    float*          he     = (float*)(ws + OFF_HE);
    float*          hd     = (float*)(ws + OFF_HD);
    unsigned short* w1e_hi = (unsigned short*)(ws + OFF_W1E_HI);
    unsigned short* w1e_lo = (unsigned short*)(ws + OFF_W1E_LO);
    unsigned short* w1d_hi = (unsigned short*)(ws + OFF_W1D_HI);
    unsigned short* w1d_lo = (unsigned short*)(ws + OFF_W1D_LO);
    unsigned short* w2_hi  = (unsigned short*)(ws + OFF_W2_HI);
    unsigned short* w2_lo  = (unsigned short*)(ws + OFF_W2_LO);

    // 1) Pre-swizzle weights into WMMA fragment layout (hi/lo bf16).
    {
        int tot = NT_K * KSTEPS * 32;
        prep_b_frags<<<(tot + 255) / 256, 256, 0, stream>>>(W1, 2 * DIM_D, 0,     DIM_K, NT_K, w1e_hi, w1e_lo);
        prep_b_frags<<<(tot + 255) / 256, 256, 0, stream>>>(W1, 2 * DIM_D, DIM_D, DIM_K, NT_K, w1d_hi, w1d_lo);
        int tot2 = NT_V * KSTEPS * 32;
        prep_b_frags<<<(tot2 + 255) / 256, 256, 0, stream>>>(W2, DIM_K, 0, DIM_V, NT_V, w2_hi, w2_lo);
    }

    // 2) he = enc @ W1e^T ; hd = dec @ W1d^T + b1  (b1 folded here)
    gemm_h_stage<<<(DIM_B * DIM_T) / 16, 256, 0, stream>>>(
        enc, (const v16bf*)w1e_hi, (const v16bf*)w1e_lo, nullptr, he);
    gemm_h_stage<<<(DIM_B * DIM_U) / 16, 256, 0, stream>>>(
        dec, (const v16bf*)w1d_hi, (const v16bf*)w1d_lo, b1, hd);

    // 3) Fused tanh + V-projection, 64 rows (full U) per block.
    joint_kernel<<<DIM_B * DIM_T, 256, 0, stream>>>(
        he, hd, (const v16bf*)w2_hi, (const v16bf*)w2_lo, b2, out);
}
